// SmartPoolSlowCNN_15917148799150
// MI455X (gfx1250) — matrix operand order repaired and checked by reading, not compiled, not run on hardware
//
#include <hip/hip_runtime.h>
#include <hip/hip_bf16.h>

typedef __attribute__((ext_vector_type(16))) _Float16 v16h;
typedef __attribute__((ext_vector_type(8)))  float    v8f;

#define THRESH 10.0f

__device__ __forceinline__ float smart_pool4(float x0, float x1, float x2, float x3) {
    // mn over the raw 2x2 window; elements with |x|>T replaced by mn; then max.
    float mn = fminf(fminf(x0, x1), fminf(x2, x3));
    float a0 = (fabsf(x0) > THRESH) ? mn : x0;
    float a1 = (fabsf(x1) > THRESH) ? mn : x1;
    float a2 = (fabsf(x2) > THRESH) ? mn : x2;
    float a3 = (fabsf(x3) > THRESH) ? mn : x3;
    return fmaxf(fmaxf(a0, a1), fmaxf(a2, a3));
}

// Fused 3x3 SAME conv (+bias, optional ReLU) + smart 2x2 pool via WMMA implicit GEMM.
//
// GEMM: D[M=16 spatial, N=16 cout] = A[M,K] x B[K,N], K re-indexed as
//   k = ci*16 + dy*4 + dx   (3x3 taps padded to 4x4)
// Pad-K entries (dx==3 | dy==3 | ci>=CIN) hold exact zeros in B, so the
// A-side gather needs only the SPATIAL bounds test; tap/channel validity is
// absorbed by the GEMM (0 * finite == 0). Gathers are branchless
// (clamped address -> unconditional load -> cndmask) and K is looped in
// chunks of 32 with v_wmma_f32_16x16x32_f16 (f32 accumulate).
//
// B fragments (weights, f16, ISA B-matrix layout) are prebuilt once per block
// into LDS; the per-chunk B read is one 32B LDS vector load per lane.
//
// M = 4x4 conv-output patch (m = py*4+px): with the ISA C/D layout each lane's
// 8 accumulators contain complete 2x2 pool windows -> pooling is in-lane VALU.
//
// TIN/TOUT: inter-stage activations are stored f16 (WMMA eats f16 anyway; this
// removes per-element cvts in the next layer's gather and halves bandwidth).
template <typename TIN, typename TOUT, int CIN, int COUT, int H, int W, bool RELU>
__global__ void conv3x3_relu_smartpool_wmma(const TIN* __restrict__ in,
                                            const float* __restrict__ wgt,
                                            const float* __restrict__ bias,
                                            TOUT* __restrict__ out,
                                            int N) {
    constexpr int KTOT   = CIN * 16;               // padded-K extent
    constexpr int NCHUNK = (KTOT + 31) / 32;       // 2 / 4 / 8
    constexpr int TX     = W / 4;
    constexpr int TY     = H / 4;
    constexpr int NTILES = (COUT + 15) / 16;
    constexpr int TILES_PER_IMG = TX * TY * NTILES;
    constexpr int PH = H / 2, PW = W / 2;

    // ---- Phase 1: build f16 B-fragments in LDS (all threads participate) ----
    // bfrag[((nt*NCHUNK + ch)*32 + lane)*16 + i] == B[k, n] for
//   n = (lane&15) + nt*16,  k = ch*32 + (i<8 ? i : i+8) + 8*(lane>=16)   (ISA B layout)
    constexpr int BTOT = NTILES * NCHUNK * 32 * 16;
    __shared__ _Float16 bfrag[BTOT];
    for (int idx = threadIdx.x; idx < BTOT; idx += blockDim.x) {
        const int i    = idx & 15;
        const int ln   = (idx >> 4) & 31;
        const int rest = idx >> 9;                 // nt*NCHUNK + ch
        const int ch   = rest % NCHUNK;
        const int nt   = rest / NCHUNK;
        const int nch  = (ln & 15) + nt * 16;
        const int kb   = ch * 32 + ((i < 8) ? i : i + 8) + ((ln >> 4) << 3);
        const int ci   = kb >> 4;
        const int dy   = (kb >> 2) & 3;
        const int dx   = kb & 3;
        const bool valid = (dx < 3) & (dy < 3) & (ci < CIN) & (nch < COUT);
        const int wi = min(nch, COUT - 1) * (CIN * 9)
                     + min(ci, CIN - 1) * 9 + min(dy, 2) * 3 + min(dx, 2);
        const float wv = wgt[wi];                  // unconditional (clamped idx)
        bfrag[idx] = (_Float16)(valid ? wv : 0.0f);
    }
    __syncthreads();

    // ---- Phase 2: per-wave WMMA tile (wave-uniform guard, EXEC all-ones) ----
    const int wid  = blockIdx.x * (blockDim.x >> 5) + (threadIdx.x >> 5);
    const int lane = threadIdx.x & 31;

    if (wid < N * TILES_PER_IMG) {
        const int n  = wid / TILES_PER_IMG;
        int r        = wid % TILES_PER_IMG;
        const int nt = r / (TX * TY);
        r            = r % (TX * TY);
        const int ty = r / TX;
        const int tx = r % TX;

        const int m     = lane & 15;               // A: both half-waves M=0..15
        const int py    = m >> 2;
        const int px    = m & 3;
        const int hiK8  = (lane >> 4) << 3;        // +8 K-offset, upper half (A)
        const int nch   = (lane & 15) + nt * 16;
        const int ybase = ty * 4 + py - 1;         // iy = ybase + dy
        const int xbase = tx * 4 + px - 1;

        const TIN*  inN = in + (long)n * CIN * H * W;
        const v16h* bfv = (const v16h*)bfrag;

        v8f acc = {};
#pragma unroll
        for (int ch = 0; ch < NCHUNK; ++ch) {
            v16h a;
#pragma unroll
            for (int i = 0; i < 16; ++i) {
                // ISA 16-bit A 16x32 layout: element i covers k-offset
                // (i<8 ? i : i+8) + 8*(lane>=16) within this 32-wide chunk.
                const int ka = ch * 32 + ((i < 8) ? i : i + 8) + hiK8;
                const int ci = ka >> 4;
                const int dy = (ka >> 2) & 3;
                const int dx = ka & 3;
                const int iy = ybase + dy;
                const int ix = xbase + dx;
                // spatial validity only; pad-K entries are zeroed via B
                const bool valid = ((unsigned)iy < (unsigned)H) &
                                   ((unsigned)ix < (unsigned)W);
                const int iyc = min(max(iy, 0), H - 1);
                const int ixc = min(max(ix, 0), W - 1);
                const int cic = (CIN & (CIN - 1)) ? min(ci, CIN - 1) : ci;
                const _Float16 t = (_Float16)inN[(cic * H + iyc) * W + ixc];
                a[i] = valid ? t : (_Float16)0;
            }
            const v16h b = bfv[(nt * NCHUNK + ch) * 32 + lane]; // 2x ds_load_b128
            acc = __builtin_amdgcn_wmma_f32_16x16x32_f16(
                /*neg_a=*/false, a, /*neg_b=*/false, b,
                /*c_mod=*/(short)0, acc, /*reuse_a=*/false, /*reuse_b=*/false);
        }

        // Bias + (ReLU) + in-lane smart pool of the two 2x2 windows this lane owns.
        const float bs = bias[min(nch, COUT - 1)];
        float v[8];
#pragma unroll
        for (int i = 0; i < 8; ++i) {
            float t = acc[i] + bs;
            v[i] = RELU ? fmaxf(t, 0.0f) : t;
        }
        const float p0 = smart_pool4(v[0], v[1], v[4], v[5]);  // window qx=0
        const float p1 = smart_pool4(v[2], v[3], v[6], v[7]);  // window qx=1

        if (COUT >= 16 || nch < COUT) {            // compile-time true for L2/L3
            const int qy = 2 * ty + (lane >> 4);   // lanes 0-15: rows 0-1; 16-31: 2-3
            TOUT* o = out + (((long)n * COUT + nch) * PH + qy) * PW + 2 * tx;
            o[0] = (TOUT)p0;
            o[1] = (TOUT)p1;
        }
    }
}

// Plain smart 2x2 pool (no conv), [N,C,H,W] f32 -> [N,C,H/2,W/2] f32
template <int C, int H, int W>
__global__ void smartpool_only(const float* __restrict__ in, float* __restrict__ out, int N) {
    constexpr int PH = H / 2, PW = W / 2;
    const int total = N * C * PH * PW;
    int idx = blockIdx.x * blockDim.x + threadIdx.x;
    if (idx >= total) return;
    const int qx = idx % PW;
    const int qy = (idx / PW) % PH;
    const int c  = (idx / (PW * PH)) % C;
    const int n  = idx / (PW * PH * C);
    const float* p = in + (((long)n * C + c) * H + 2 * qy) * W + 2 * qx;
    out[idx] = smart_pool4(p[0], p[1], p[W], p[W + 1]);
}

// FC: out[n][o] = sum_j a[n*800+j] * fcw[o*800+j] + fcb[o]; tiny (N*2 outputs).
__global__ void fc_kernel(const float* __restrict__ a,
                          const float* __restrict__ fcw,
                          const float* __restrict__ fcb,
                          float* __restrict__ out, int N) {
    int idx = blockIdx.x * blockDim.x + threadIdx.x;
    if (idx >= N * 2) return;
    const int n = idx >> 1;
    const int o = idx & 1;
    const float* av = a + (long)n * 800;
    const float* wv = fcw + o * 800;
    float s = fcb[o];
#pragma unroll 8
    for (int j = 0; j < 800; ++j) s = fmaf(av[j], wv[j], s);
    out[idx] = s;
}

extern "C" void kernel_launch(void* const* d_in, const int* in_sizes, int n_in,
                              void* d_out, int out_size, void* d_ws, size_t ws_size,
                              hipStream_t stream) {
    const float* x   = (const float*)d_in[0];
    const float* w1  = (const float*)d_in[1];
    const float* b1  = (const float*)d_in[2];
    const float* w2  = (const float*)d_in[3];
    const float* b2  = (const float*)d_in[4];
    const float* w3  = (const float*)d_in[5];
    const float* b3  = (const float*)d_in[6];
    const float* fcw = (const float*)d_in[7];
    const float* fcb = (const float*)d_in[8];
    float* out = (float*)d_out;

    const int N = in_sizes[0] / (3 * 80 * 80);   // 512

    // Workspace carve-up (f16 inter-stage activations for WMMA consumers).
    char* ws = (char*)d_ws;
    _Float16* a1 = (_Float16*)ws;                              // [N, 8,40,40] f16
    ws += (size_t)N * 8 * 40 * 40 * sizeof(_Float16);
    _Float16* a2 = (_Float16*)ws;                              // [N,16,20,20] f16
    ws += (size_t)N * 16 * 20 * 20 * sizeof(_Float16);
    float* a3 = (float*)ws;                                    // [N,32,10,10] f32
    ws += (size_t)N * 32 * 10 * 10 * sizeof(float);
    float* a4 = (float*)ws;                                    // [N,32, 5, 5] f32

    const int block = 128;                        // 4 waves/block
    const int wpb   = block / 32;

    // L1: conv 3->8 on 80x80 (f32 in, f16 out). Waves: N*20*20
    {
        int waves = N * 20 * 20 * 1;
        conv3x3_relu_smartpool_wmma<float, _Float16, 3, 8, 80, 80, true>
            <<<(waves + wpb - 1) / wpb, block, 0, stream>>>(x, w1, b1, a1, N);
    }
    // L2: conv 8->16 on 40x40 (f16 in, f16 out). Waves: N*10*10
    {
        int waves = N * 10 * 10 * 1;
        conv3x3_relu_smartpool_wmma<_Float16, _Float16, 8, 16, 40, 40, true>
            <<<(waves + wpb - 1) / wpb, block, 0, stream>>>(a1, w2, b2, a2, N);
    }
    // L3: conv 16->32 on 20x20 (f16 in, f32 out). Waves: N*5*5*2
    {
        int waves = N * 5 * 5 * 2;
        conv3x3_relu_smartpool_wmma<_Float16, float, 16, 32, 20, 20, true>
            <<<(waves + wpb - 1) / wpb, block, 0, stream>>>(a2, w3, b3, a3, N);
    }
    // L4: smart pool only, [N,32,10,10] -> [N,32,5,5]
    {
        int total = N * 32 * 5 * 5;
        smartpool_only<32, 10, 10><<<(total + 255) / 256, 256, 0, stream>>>(a3, a4, N);
    }
    // FC: [N,800] @ [800,2]^T + b -> d_out
    {
        int total = N * 2;
        fc_kernel<<<(total + 255) / 256, 256, 0, stream>>>(a4, fcw, fcb, out, N);
    }
}
